// BahdanauAttention_42021960024523
// MI455X (gfx1250) — matrix operand order repaired and checked by reading, not compiled, not run on hardware
//
#include <hip/hip_runtime.h>
#include <hip/hip_bf16.h>
#include <math.h>

// Bahdanau attention, MI455X (gfx1250), wave32 + WMMA bf16.
// B=32, S=2048, D=1024, U=1024.

typedef __attribute__((ext_vector_type(8)))  __bf16 v8bf;
typedef __attribute__((ext_vector_type(16))) __bf16 v16bf;
typedef __attribute__((ext_vector_type(8)))  float  v8f;

#define BB 32
#define SS 2048
#define DD 1024
#define UU 1024
#define MT 32   // S-rows per workgroup in the score kernel

// ---------------------------------------------------------------------------
// Kernel A: pack w2 (fp32 [D,U]) into bf16 WMMA-B fragment order.
// Packed index p = ((kt*64 + ntg)*32 + lane)*16 + j
//   lanes 0-15 : K = kt*32 + j      , N = ntg*16 + lane
//   lanes 16-31: K = kt*32 + 16 + j , N = ntg*16 + (lane-16)
// ---------------------------------------------------------------------------
__global__ __launch_bounds__(256) void pack_w2_kernel(
    const float* __restrict__ w2, __bf16* __restrict__ w2p)
{
  int p = blockIdx.x * 256 + threadIdx.x;      // 0 .. 1M-1
  int j    = p & 15;
  int lane = (p >> 4) & 31;
  int ntg  = (p >> 9) & 63;
  int kt   = p >> 15;                          // 0..31
  int K = kt * 32 + ((lane >> 4) << 4) + j;
  int n = ntg * 16 + (lane & 15);
  w2p[p] = (__bf16)w2[(size_t)K * UU + n];
}

// ---------------------------------------------------------------------------
// Kernel B: qpb2[b,u] = query[b,:]@w1[:,u] + b1[u] + b2[u]
// ---------------------------------------------------------------------------
__global__ __launch_bounds__(256) void qproj_kernel(
    const float* __restrict__ query, const float* __restrict__ w1,
    const float* __restrict__ b1, const float* __restrict__ b2,
    float* __restrict__ qpb2)
{
  int idx = blockIdx.x * 256 + threadIdx.x;    // 0 .. 32*1024-1
  int b = idx >> 10, u = idx & 1023;
  const float* qr = query + (size_t)b * DD;
  float acc = 0.f;
  for (int d = 0; d < DD; ++d)
    acc = fmaf(qr[d], w1[(size_t)d * UU + u], acc);
  qpb2[idx] = acc + b1[u] + b2[u];
}

// ---------------------------------------------------------------------------
// Kernel C (hot): score[b,s] = sum_u tanh(qpb2[b,u] + values[b,s,:]@w2[:,u]) * v[u] + bv
// One WG = (b, 32-row s-tile). values tile staged in LDS as bf16 in WMMA-A
// fragment order; w2 read pre-packed. 8 waves tile C 32x64; K-loop of WMMAs.
// ---------------------------------------------------------------------------
__global__ __launch_bounds__(256) void bahdanau_score_kernel(
    const float* __restrict__ values, const __bf16* __restrict__ w2p,
    const float* __restrict__ qpb2, const float* __restrict__ vvec,
    const float* __restrict__ bvp, float* __restrict__ score)
{
  extern __shared__ char smem[];
  __bf16* aLds = (__bf16*)smem;                    // [2 mt][32 kt][32 lane][16] = 64 KB
  float*  wsum = (float*)(smem + MT * DD * 2);     // [8 waves][16 rows]

  const int blk = blockIdx.x;
  const int b   = blk >> 6;                        // 64 s-tiles per batch
  const int s0  = (blk & 63) * MT;
  const int tid = threadIdx.x;

  // ---- stage values[b, s0:s0+32, :] -> LDS bf16, A-fragment order ----
  const float* vbase = values + ((size_t)b * SS + s0) * DD;
  #pragma unroll
  for (int i = 0; i < 16; ++i) {
    int g  = tid + i * 256;                        // chunk of 8 consecutive floats
    int r  = g >> 7;                               // row 0..31
    int kc = g & 127;                              // 8-float chunk along K
    const float4* src = (const float4*)(vbase + (size_t)r * DD + kc * 8);
    float4 a0 = src[0], a1 = src[1];
    int mt   = r >> 4, l15 = r & 15;
    int kt   = kc >> 2;
    int koff = (kc & 3) * 8;                       // 0,8,16,24
    int ln   = l15 + ((koff & 16) ? 16 : 0);
    int half = (koff >> 3) & 1;
    int off  = (((mt * 32 + kt) * 32 + ln) * 2 + half) * 8;
    v8bf pk;
    pk[0]=(__bf16)a0.x; pk[1]=(__bf16)a0.y; pk[2]=(__bf16)a0.z; pk[3]=(__bf16)a0.w;
    pk[4]=(__bf16)a1.x; pk[5]=(__bf16)a1.y; pk[6]=(__bf16)a1.z; pk[7]=(__bf16)a1.w;
    *(v8bf*)(aLds + off) = pk;
  }
  __syncthreads();

  const int wave = tid >> 5;
  const int lane = tid & 31;
  const int mt   = wave & 1;                       // 16-row tile within 32 rows
  const int ntw  = wave >> 1;                      // 0..3: 16-col tile within 64-col chunk

  float acc[8] = {0.f,0.f,0.f,0.f,0.f,0.f,0.f,0.f};
  const float* qrow = qpb2 + (size_t)b * UU;

  for (int nIter = 0; nIter < UU / 64; ++nIter) {
    const int ntg = nIter * 4 + ntw;               // global 16-col tile, 0..63
    v8f c = {};
    #pragma unroll 4
    for (int kt = 0; kt < DD / 32; ++kt) {
      const v8bf* ap = (const v8bf*)(aLds + (size_t)(((mt * 32 + kt) * 32 + lane) * 2) * 8);
      v8bf a0 = ap[0], a1 = ap[1];
      const v8bf* bp = (const v8bf*)(w2p + (((size_t)kt * 64 + ntg) * 32 + lane) * 16);
      v8bf b0 = bp[0], b1 = bp[1];
      v16bf av  = __builtin_shufflevector(a0, a1, 0,1,2,3,4,5,6,7,8,9,10,11,12,13,14,15);
      v16bf bmv = __builtin_shufflevector(b0, b1, 0,1,2,3,4,5,6,7,8,9,10,11,12,13,14,15);
      c = __builtin_amdgcn_wmma_f32_16x16x32_bf16(false, av, false, bmv,
                                                  (short)0, c, false, false);
    }
    // C layout: reg r -> (M = r + 8*(lane>=16), N = lane&15); apply tanh * v[u]
    const int   n  = ntg * 16 + (lane & 15);
    const float q  = qrow[n];
    const float vw = vvec[n];
    #pragma unroll
    for (int r = 0; r < 8; ++r)
      acc[r] += tanhf(c[r] + q) * vw;
  }

  // reduce over the 16 N-lanes within each half-wave (xor 8,4,2,1)
  #pragma unroll
  for (int r = 0; r < 8; ++r) {
    float x = acc[r];
    x += __shfl_xor(x, 8, 32);
    x += __shfl_xor(x, 4, 32);
    x += __shfl_xor(x, 2, 32);
    x += __shfl_xor(x, 1, 32);
    acc[r] = x;
  }
  if ((lane & 15) == 0) {
    const int rbase = (lane >> 4) * 8;             // lane 0 -> rows 0..7, lane 16 -> rows 8..15
    #pragma unroll
    for (int r = 0; r < 8; ++r) wsum[wave * 16 + rbase + r] = acc[r];
  }
  __syncthreads();
  if (tid < MT) {                                  // deterministic fixed-order combine
    const int mt2 = tid >> 4, rl = tid & 15;
    float s = bvp[0];
    #pragma unroll
    for (int nt = 0; nt < 4; ++nt) s += wsum[(nt * 2 + mt2) * 16 + rl];
    score[(size_t)b * SS + s0 + tid] = s;
  }
}

// ---------------------------------------------------------------------------
// Kernel D: masked softmax over S, writes attention weights.
// ---------------------------------------------------------------------------
__global__ __launch_bounds__(256) void softmax_kernel(
    const float* __restrict__ score, const int* __restrict__ mask,
    float* __restrict__ attn)
{
  __shared__ float red[256];
  const int b = blockIdx.x, tid = threadIdx.x;
  float sc[8];
  float mx = -INFINITY;
  #pragma unroll
  for (int i = 0; i < 8; ++i) {
    int s = tid + i * 256;
    float x = score[(size_t)b * SS + s];
    x += (1.0f - (float)mask[(size_t)b * SS + s]) * -1.0e9f;
    sc[i] = x;
    mx = fmaxf(mx, x);
  }
  red[tid] = mx; __syncthreads();
  for (int o = 128; o > 0; o >>= 1) {
    if (tid < o) red[tid] = fmaxf(red[tid], red[tid + o]);
    __syncthreads();
  }
  mx = red[0]; __syncthreads();
  float lsum = 0.f;
  #pragma unroll
  for (int i = 0; i < 8; ++i) { sc[i] = expf(sc[i] - mx); lsum += sc[i]; }
  red[tid] = lsum; __syncthreads();
  for (int o = 128; o > 0; o >>= 1) {
    if (tid < o) red[tid] += red[tid + o];
    __syncthreads();
  }
  const float inv = 1.0f / red[0];
  #pragma unroll
  for (int i = 0; i < 8; ++i)
    attn[(size_t)b * SS + tid + i * 256] = sc[i] * inv;
}

// ---------------------------------------------------------------------------
// Kernel E: context partials: part[sc][b][d] = sum over 128 s of attn*values
// ---------------------------------------------------------------------------
__global__ __launch_bounds__(256) void ctx_partial_kernel(
    const float* __restrict__ values, const float* __restrict__ attn,
    float* __restrict__ part)
{
  const int b = blockIdx.x, sc = blockIdx.y;
  const int d4 = threadIdx.x * 4;
  const float* vb = values + ((size_t)b * SS + sc * 128) * DD;
  const float* ab = attn + (size_t)b * SS + sc * 128;
  float4 acc = {0.f, 0.f, 0.f, 0.f};
  for (int i = 0; i < 128; ++i) {
    float a = ab[i];
    const float4 vv = *(const float4*)(vb + (size_t)i * DD + d4);
    acc.x = fmaf(a, vv.x, acc.x);
    acc.y = fmaf(a, vv.y, acc.y);
    acc.z = fmaf(a, vv.z, acc.z);
    acc.w = fmaf(a, vv.w, acc.w);
  }
  *(float4*)(part + ((size_t)sc * BB + b) * DD + d4) = acc;
}

// ---------------------------------------------------------------------------
// Kernel F: deterministic reduce of context partials -> d_out[0:32768]
// ---------------------------------------------------------------------------
__global__ __launch_bounds__(256) void ctx_reduce_kernel(
    const float* __restrict__ part, float* __restrict__ outc)
{
  int idx = blockIdx.x * 256 + threadIdx.x;      // 0..32767
  float acc = 0.f;
  #pragma unroll
  for (int sc = 0; sc < 16; ++sc) acc += part[(size_t)sc * (BB * DD) + idx];
  outc[idx] = acc;
}

// ---------------------------------------------------------------------------
extern "C" void kernel_launch(void* const* d_in, const int* in_sizes, int n_in,
                              void* d_out, int out_size, void* d_ws, size_t ws_size,
                              hipStream_t stream) {
  const float* query  = (const float*)d_in[0];   // [32,1024]
  const float* values = (const float*)d_in[1];   // [32,2048,1024]
  const int*   mask   = (const int*)  d_in[2];   // [32,2048]
  const float* w1     = (const float*)d_in[3];   // [1024,1024]
  const float* b1     = (const float*)d_in[4];   // [1024]
  const float* w2     = (const float*)d_in[5];   // [1024,1024]
  const float* b2     = (const float*)d_in[6];   // [1024]
  const float* vvec   = (const float*)d_in[7];   // [1024,1]
  const float* bv     = (const float*)d_in[8];   // [1]

  float* out  = (float*)d_out;                   // [0:32768) context, [32768:98304) attn
  char*  ws   = (char*)d_ws;
  __bf16* w2p  = (__bf16*)ws;                                    // 2 MB
  float*  qpb2 = (float*)(ws + (2u << 20));                      // 128 KB
  float*  scor = (float*)(ws + (2u << 20) + (128u << 10));       // 256 KB
  float*  part = (float*)(ws + (2u << 20) + (384u << 10));       // 2 MB

  float* attn_out = out + BB * DD;

  pack_w2_kernel<<<(DD * UU) / 256, 256, 0, stream>>>(w2, w2p);
  qproj_kernel<<<(BB * UU) / 256, 256, 0, stream>>>(query, w1, b1, b2, qpb2);

  const size_t smem = (size_t)MT * DD * 2 + 8 * 16 * sizeof(float); // 66048 B
  bahdanau_score_kernel<<<BB * (SS / MT), 256, smem, stream>>>(
      values, w2p, qpb2, vvec, bv, scor);

  softmax_kernel<<<BB, 256, 0, stream>>>(scor, mask, attn_out);
  ctx_partial_kernel<<<dim3(BB, 16), 256, 0, stream>>>(values, attn_out, part);
  ctx_reduce_kernel<<<(BB * DD) / 256, 256, 0, stream>>>(part, out);
}
